// _SimpleBlock2d_55018531062098
// MI455X (gfx1250) — matrix operand order repaired and checked by reading, not compiled, not run on hardware
//
#include <hip/hip_runtime.h>
#include <hip/hip_bf16.h>

// ---------------------------------------------------------------------------
// FNO SimpleBlock2d for MI455X (gfx1250, wave32, WMMA).
// Truncated-DFT-as-GEMM formulation; all matrix math via
// v_wmma_f32_16x16x32_f16. All operand zero-padding lives in the data
// layout (padded tables / strides) so fragment gathers are branch-free.
// ---------------------------------------------------------------------------

typedef __attribute__((ext_vector_type(16))) _Float16 v16h;
typedef __attribute__((ext_vector_type(8)))  float    v8f;

// neg_a/neg_b must be 0 for f16 WMMA (ISA: NEG = {CNeg,0,0} only).
#define WMMA(a, b, c) __builtin_amdgcn_wmma_f32_16x16x32_f16(false, (a), false, (b), (short)0, (c), false, false)

// Problem constants
#define NB   16        // batch
#define NC   32        // width (channels)
#define NH   255       // mirrored spatial size (2*128-1)
#define NS   128       // original spatial size
#define NM1  12        // modes1
#define NM2  12        // modes2
#define NK1  24        // kept k1 values: 0..11 and 243..254
#define HW   65025     // logical pixels per plane (255*255)
#define PL   65280     // padded plane: 255 rows * 256 cols
#define CHW2 2088960   // 32 * PL
#define BNEPS 1e-5f
#define TWOPI 6.28318530717958647692f

// padded intermediate strides
#define YF_STR 32      // Yf row stride (24 used, 25..31 zero by construction)
#define ZV_STR 16      // Zf/Zm col stride (12 used)
#define YI_ROWS 256    // Yi rows per (b,o): 255 + 1 pad row

// ---- fragment index helpers (16-bit A 16x32 / B 32x16; contiguous k = 16*half + e, same for A and B) --------
__device__ __forceinline__ int fragA_k(int e, int half) { return 16 * half + e; }
__device__ __forceinline__ int fragB_k(int e, int half) { return 16 * half + e; }

__device__ __forceinline__ v8f vzero8() {
    v8f z;
#pragma unroll
    for (int i = 0; i < 8; ++i) z[i] = 0.0f;
    return z;
}

__device__ __forceinline__ int k1_of(int j) { return (j < NM1) ? j : (j + 231); } // 243..254

// ---------------------------------------------------------------------------
// K0: padded DFT twiddle tables (zeros included in the padding):
//   DWF [256][32] : forward W-DFT (cos,-sin)/255; row 255 = 0, cols>=24 = 0
//   DHF [48][512] : forward H-DFT rows 2j=Re/2j+1=Im, K=2h+q; cols>=510 = 0
//   DHI [256][64] : inverse H-DFT (cos,sin); cols>=48 = 0, row 255 = 0
//   DWI [32][256] : inverse W irfft (x2 conj-sym, 1/255); rows>=24 = 0, col 255 = 0
// ---------------------------------------------------------------------------
__global__ void fno_init_tables(float* DWF, float* DHF, float* DHI, float* DWI) {
    int stride = blockDim.x * gridDim.x;
    int tid = blockIdx.x * blockDim.x + threadIdx.x;
    for (int i = tid; i < 256 * 32; i += stride) {           // DWF
        int w = i >> 5, r = i & 31;
        int k = r >> 1, p = r & 1;
        float v = 0.0f;
        if (w < NH && r < 2 * NM2) {
            float th = TWOPI * (float)((w * k) % NH) / (float)NH;
            v = (p == 0 ? cosf(th) : -sinf(th)) * (1.0f / 255.0f);
        }
        DWF[i] = v;
    }
    for (int i = tid; i < 48 * 512; i += stride) {           // DHF
        int row = i >> 9, K = i & 511;
        int j = row >> 1, pr = row & 1;
        int h = K >> 1, q = K & 1;
        float v = 0.0f;
        if (K < 2 * NH) {
            float th = TWOPI * (float)((h * k1_of(j)) % NH) / (float)NH;
            float c = cosf(th), s = sinf(th);
            v = (pr == 0) ? (q == 0 ? c : s) : (q == 0 ? -s : c);
        }
        DHF[i] = v;
    }
    for (int i = tid; i < 256 * 64; i += stride) {           // DHI
        int h = i >> 6, r = i & 63;
        int j = r >> 1, p = r & 1;
        float v = 0.0f;
        if (h < NH && r < 2 * NK1) {
            float th = TWOPI * (float)((h * k1_of(j)) % NH) / (float)NH;
            v = (p == 0) ? cosf(th) : sinf(th);
        }
        DHI[i] = v;
    }
    for (int i = tid; i < 32 * 256; i += stride) {           // DWI
        int r = i >> 8, w = i & 255;
        int k = r >> 1, p = r & 1;
        float v = 0.0f;
        if (r < 2 * NM2 && w < NH) {
            float th = TWOPI * (float)((w * k) % NH) / (float)NH;
            float cc = (k == 0 ? 1.0f : 2.0f) / 255.0f;
            v = (p == 0) ? cc * cosf(th) : -cc * sinf(th);
        }
        DWI[i] = v;
    }
}

// ---------------------------------------------------------------------------
// K1: odd mirror + fc0 lift (3 -> 32 channels), output (B,32,255,255) padded
// ---------------------------------------------------------------------------
__global__ void fno_mirror_fc0(const float* __restrict__ x,
                               const float* __restrict__ w,
                               const float* __restrict__ b,
                               float* __restrict__ outA) {
    int stride = blockDim.x * gridDim.x;
    for (int idx = blockIdx.x * blockDim.x + threadIdx.x; idx < NB * HW; idx += stride) {
        int bb = idx / HW, hw = idx - bb * HW;
        int i = hw / NH, j = hw - i * NH;
        int ri = (i < NS - 1) ? (NS - 1 - i) : (i - (NS - 1));
        int rj = (j < NS - 1) ? (NS - 1 - j) : (j - (NS - 1));
        float sgn = ((i < NS - 1) == (j < NS - 1)) ? 1.0f : -1.0f;
        const float* xb = x + (size_t)bb * 3 * NS * NS + ri * NS + rj;
        float x0 = sgn * xb[0];
        float x1 = sgn * xb[NS * NS];
        float x2 = sgn * xb[2 * NS * NS];
        float* o = outA + (size_t)bb * CHW2 + i * 256 + j;
#pragma unroll
        for (int d = 0; d < NC; ++d)
            o[d * PL] = x0 * w[d] + x1 * w[NC + d] + x2 * w[2 * NC + d] + b[d];
    }
}

// ---------------------------------------------------------------------------
// K2: forward partial DFT along W. rows = B*C*255, K=256 (col 255 pad), N=24.
//     Yf[row][32]: cols 0..23 Re/Im pairs, 24..31 exact zeros (from DWF pad).
// ---------------------------------------------------------------------------
__global__ void fno_dft_fwd_w(const float* __restrict__ A,
                              const float* __restrict__ DWF,
                              float* __restrict__ Yf) {
    int lane = threadIdx.x & 31;
    int m = lane & 15, half = lane >> 4, n = lane & 15;
    int r0 = blockIdx.x * 16;
    int flat = r0 + m;
    int bc = flat / NH, h = flat - bc * NH;
    const float* Arow = A + (size_t)bc * PL + h * 256;

    v8f c0 = vzero8(), c1 = vzero8();
#pragma unroll
    for (int kc = 0; kc < 8; ++kc) {
        v16h a, b0, b1;
#pragma unroll
        for (int e = 0; e < 16; ++e) {
            a[e] = (_Float16)Arow[kc * 32 + fragA_k(e, half)];
            int kb = kc * 32 + fragB_k(e, half);
            b0[e] = (_Float16)DWF[kb * 32 + n];
            b1[e] = (_Float16)DWF[kb * 32 + 16 + n];
        }
        c0 = WMMA(a, b0, c0);
        c1 = WMMA(a, b1, c1);
    }
#pragma unroll
    for (int r = 0; r < 8; ++r) {
        int row = r0 + r + 8 * half;
        Yf[(size_t)row * YF_STR + n] = c0[r];
        Yf[(size_t)row * YF_STR + 16 + n] = c1[r];
    }
}

// ---------------------------------------------------------------------------
// K3: forward partial DFT along H. Per (b,c): M=48, K=512 (pad), N=16 (12 used).
//     Zf[(bc*48+row)*16 + k2]
// ---------------------------------------------------------------------------
__global__ void fno_dft_fwd_h(const float* __restrict__ Yf,
                              const float* __restrict__ DHF,
                              float* __restrict__ Zf) {
    int lane = threadIdx.x & 31;
    int m = lane & 15, half = lane >> 4, n = lane & 15;
    int bc = blockIdx.x;
    v8f acc[3];
#pragma unroll
    for (int t = 0; t < 3; ++t) acc[t] = vzero8();
    for (int kc = 0; kc < 16; ++kc) {
        v16h bfr;
#pragma unroll
        for (int e = 0; e < 16; ++e) {
            int K = kc * 32 + fragB_k(e, half);
            int h = K >> 1, p = K & 1;
            // K>=510 reads cross-row garbage -> killed by DHF zero cols.
            // n>=12 reads Yf cols 24..31 == 0 -> exact zeros.
            bfr[e] = (_Float16)Yf[((size_t)bc * NH + h) * YF_STR + 2 * n + p];
        }
#pragma unroll
        for (int mt = 0; mt < 3; ++mt) {
            v16h a;
#pragma unroll
            for (int e = 0; e < 16; ++e) {
                int K = kc * 32 + fragA_k(e, half);
                a[e] = (_Float16)DHF[(size_t)(mt * 16 + m) * 512 + K];
            }
            acc[mt] = WMMA(a, bfr, acc[mt]);
        }
    }
#pragma unroll
    for (int mt = 0; mt < 3; ++mt)
#pragma unroll
        for (int r = 0; r < 8; ++r) {
            int row = mt * 16 + r + 8 * half;
            Zf[((size_t)bc * 2 * NK1 + row) * ZV_STR + n] = acc[mt][r];
        }
}

// ---------------------------------------------------------------------------
// K4: per-mode complex channel mixing. grid = 24*12 modes. M=B=16, K=C=32, N=O=32.
//     Re = Zr*Wr + (-Zi)*Wi ; Im = Zr*Wi + Zi*Wr
// ---------------------------------------------------------------------------
__global__ void fno_mix(const float* __restrict__ Zf,
                        const float* __restrict__ spec_w,
                        float* __restrict__ Zm, int blk) {
    int lane = threadIdx.x & 31;
    int m = lane & 15, half = lane >> 4, n = lane & 15;
    int j = blockIdx.x / NM2, k2 = blockIdx.x - j * NM2;
    int cr = (j < NM1) ? 0 : 1;
    int m1 = (j < NM1) ? j : (j - NM1);

    v16h Zr, Zi, Zin;   // Zin = -Zi
#pragma unroll
    for (int e = 0; e < 16; ++e) {
        int i = fragA_k(e, half);
        size_t base = ((size_t)m * NC + i) * (2 * NK1) + 2 * j;
        float zr = Zf[base * ZV_STR + k2];
        float zi = Zf[(base + 1) * ZV_STR + k2];
        Zr[e]  = (_Float16)zr;
        Zi[e]  = (_Float16)zi;
        Zin[e] = (_Float16)(-zi);
    }
#pragma unroll
    for (int nt = 0; nt < 2; ++nt) {
        v16h Wr, Wi;
#pragma unroll
        for (int e = 0; e < 16; ++e) {
            int i = fragB_k(e, half);
            int o = nt * 16 + n;
            size_t widx = (((((size_t)blk * 2 + cr) * NC + i) * NC + o) * NM1 + m1) * NM2 + k2;
            Wr[e] = (_Float16)spec_w[widx * 2 + 0];
            Wi[e] = (_Float16)spec_w[widx * 2 + 1];
        }
        v8f cRe = vzero8(), cIm = vzero8();
        cRe = WMMA(Zr, Wr, cRe);
        cRe = WMMA(Zin, Wi, cRe);   // Re -= Zi*Wi
        cIm = WMMA(Zr, Wi, cIm);
        cIm = WMMA(Zi, Wr, cIm);
#pragma unroll
        for (int r = 0; r < 8; ++r) {
            int bb = r + 8 * half;
            int o = nt * 16 + n;
            size_t base = ((size_t)bb * NC + o) * (2 * NK1) + 2 * j;
            Zm[base * ZV_STR + k2] = cRe[r];
            Zm[(base + 1) * ZV_STR + k2] = cIm[r];
        }
    }
}

// ---------------------------------------------------------------------------
// K5: inverse DFT along H. Per (b,o): M=256 (row 255 pad), K=64 (pad), N=32.
//     Yi[(bo*256+h)*32 + 2*k2+q]; cols 24..31 garbage, killed later by DWI pad.
// ---------------------------------------------------------------------------
__global__ void fno_dft_inv_h(const float* __restrict__ Zm,
                              const float* __restrict__ DHI,
                              float* __restrict__ Yi) {
    int lane = threadIdx.x & 31;
    int m = lane & 15, half = lane >> 4, n = lane & 15;
    int bo = blockIdx.x;

    v16h Bf[2][2];   // [kchunk][ntile]
#pragma unroll
    for (int kc = 0; kc < 2; ++kc)
#pragma unroll
        for (int nt = 0; nt < 2; ++nt)
#pragma unroll
            for (int e = 0; e < 16; ++e) {
                int K = kc * 32 + fragB_k(e, half);   // 0..63; >=48 killed by DHI pad
                int nn = nt * 16 + n;                 // 0..31; >=24 produces pad cols
                int j = K >> 1, p = K & 1;
                int k2 = nn >> 1, q = nn & 1;
                size_t base = (size_t)bo * (2 * NK1) + 2 * j;
                float zr = Zm[base * ZV_STR + k2];
                float zi = Zm[(base + 1) * ZV_STR + k2];
                float v = (p == 0) ? (q == 0 ? zr : zi) : (q == 0 ? -zi : zr);
                Bf[kc][nt][e] = (_Float16)v;
            }

    for (int mt = 0; mt < 16; ++mt) {
        v8f c0 = vzero8(), c1 = vzero8();
#pragma unroll
        for (int kc = 0; kc < 2; ++kc) {
            v16h a;
#pragma unroll
            for (int e = 0; e < 16; ++e) {
                int h = mt * 16 + m;                  // 0..255 (row 255 = pad row)
                int K = kc * 32 + fragA_k(e, half);
                a[e] = (_Float16)DHI[(size_t)h * 64 + K];
            }
            c0 = WMMA(a, Bf[kc][0], c0);
            c1 = WMMA(a, Bf[kc][1], c1);
        }
#pragma unroll
        for (int r = 0; r < 8; ++r) {
            int h = mt * 16 + r + 8 * half;
            Yi[((size_t)bo * YI_ROWS + h) * 32 + n] = c0[r];
            Yi[((size_t)bo * YI_ROWS + h) * 32 + 16 + n] = c1[r];
        }
    }
}

// ---------------------------------------------------------------------------
// K6: inverse irfft along W (real out). rows = B*O*255, K=32 (pad), N=256.
//     writes x1 into bufS padded planes (col 255 gets exact 0 from DWI pad).
// ---------------------------------------------------------------------------
__global__ void fno_dft_inv_w(const float* __restrict__ Yi,
                              const float* __restrict__ DWI,
                              float* __restrict__ outS) {
    int lane = threadIdx.x & 31;
    int m = lane & 15, half = lane >> 4, n = lane & 15;
    int r0 = blockIdx.x * 16;
    int flat = r0 + m;
    int bo = flat / NH, h = flat - bo * NH;

    v16h a;
#pragma unroll
    for (int e = 0; e < 16; ++e) {
        int k = fragA_k(e, half);   // 0..31; cols 24..31 garbage killed by DWI pad rows
        a[e] = (_Float16)Yi[((size_t)bo * YI_ROWS + h) * 32 + k];
    }
#pragma unroll
    for (int nt = 0; nt < 16; ++nt) {
        v16h b;
#pragma unroll
        for (int e = 0; e < 16; ++e) {
            int k = fragB_k(e, half);
            int w = nt * 16 + n;
            b[e] = (_Float16)DWI[(size_t)k * 256 + w];
        }
        v8f c = vzero8();
        c = WMMA(a, b, c);
#pragma unroll
        for (int r = 0; r < 8; ++r) {
            int flat2 = r0 + r + 8 * half;
            int bo2 = flat2 / NH, h2 = flat2 - bo2 * NH;
            int w = nt * 16 + n;
            outS[(size_t)bo2 * PL + h2 * 256 + w] = c[r];
        }
    }
}

// ---------------------------------------------------------------------------
// K7: pointwise conv 32x32 and add into bufS (which already holds x1).
// ---------------------------------------------------------------------------
__global__ void fno_conv_add(const float* __restrict__ Abuf,
                             const float* __restrict__ conv_w,
                             const float* __restrict__ conv_b,
                             float* __restrict__ Sbuf, int blk) {
    int lane = threadIdx.x & 31;
    int m = lane & 15, half = lane >> 4, n = lane & 15;
    int t0 = blockIdx.x * 16;

    int pix = t0 + m;
    int bb = pix / HW, hw = pix - bb * HW;
    int i = hw / NH, j = hw - i * NH;
    const float* abase = Abuf + (size_t)bb * CHW2 + i * 256 + j;
    __builtin_prefetch(abase + 16, 0, 1);   // global_prefetch_b8: next pixel tile
    v16h a;
#pragma unroll
    for (int e = 0; e < 16; ++e) {
        int cch = fragA_k(e, half);
        a[e] = (_Float16)abase[(size_t)cch * PL];
    }
#pragma unroll
    for (int nt = 0; nt < 2; ++nt) {
        v16h b;
#pragma unroll
        for (int e = 0; e < 16; ++e) {
            int cch = fragB_k(e, half);
            int o = nt * 16 + n;
            b[e] = (_Float16)conv_w[(size_t)blk * NC * NC + o * NC + cch];
        }
        v8f c = vzero8();
        c = WMMA(a, b, c);
#pragma unroll
        for (int r = 0; r < 8; ++r) {
            int p2 = t0 + r + 8 * half;
            int b2 = p2 / HW, hw2 = p2 - b2 * HW;
            int i2 = hw2 / NH, j2 = hw2 - i2 * NH;
            int o = nt * 16 + n;
            size_t addr = (size_t)b2 * CHW2 + (size_t)o * PL + i2 * 256 + j2;
            Sbuf[addr] += c[r] + conv_b[blk * NC + o];
        }
    }
}

// ---------------------------------------------------------------------------
// K8: zero helper and per-channel sum / sumsq reduction for BN.
// ---------------------------------------------------------------------------
__global__ void fno_zero(float* p, int nn) {
    int i = blockIdx.x * blockDim.x + threadIdx.x;
    if (i < nn) p[i] = 0.0f;
}

__global__ void fno_bn_stats(const float* __restrict__ Sbuf, float* __restrict__ stats) {
    __shared__ float ssum[256], ssq[256];
    int c = blockIdx.x >> 3;          // 32 channels * 8 parts
    int part = blockIdx.x & 7;
    float s = 0.0f, q = 0.0f;
    for (int e = part * 256 + threadIdx.x; e < NB * HW; e += 8 * 256) {
        int bb = e / HW, hw = e - bb * HW;
        int i = hw / NH, j = hw - i * NH;
        float v = Sbuf[(size_t)bb * CHW2 + (size_t)c * PL + i * 256 + j];
        s += v; q += v * v;
    }
    ssum[threadIdx.x] = s; ssq[threadIdx.x] = q;
    __syncthreads();
    for (int off = 128; off > 0; off >>= 1) {
        if (threadIdx.x < off) {
            ssum[threadIdx.x] += ssum[threadIdx.x + off];
            ssq[threadIdx.x] += ssq[threadIdx.x + off];
        }
        __syncthreads();
    }
    if (threadIdx.x == 0) {
        atomicAdd(&stats[c], ssum[0]);
        atomicAdd(&stats[NC + c], ssq[0]);
    }
}

__global__ void fno_bn_apply(const float* __restrict__ Sbuf,
                             const float* __restrict__ stats,
                             const float* __restrict__ g,
                             const float* __restrict__ bta,
                             float* __restrict__ Abuf, int blk, int dorelu) {
    const float invcnt = 1.0f / (float)(NB * HW);
    size_t stride = (size_t)blockDim.x * gridDim.x;
    for (size_t idx = (size_t)blockIdx.x * blockDim.x + threadIdx.x;
         idx < (size_t)NB * NC * HW; idx += stride) {
        int hw = (int)(idx % HW);
        int bc = (int)(idx / HW);
        int c = bc & (NC - 1);
        int i = hw / NH, j = hw - i * NH;
        size_t addr = (size_t)(bc >> 5) * CHW2 + (size_t)c * PL + i * 256 + j;
        float mean = stats[c] * invcnt;
        float var = stats[NC + c] * invcnt - mean * mean;
        float inv = rsqrtf(var + BNEPS);
        float v = g[blk * NC + c] * (Sbuf[addr] - mean) * inv + bta[blk * NC + c];
        if (dorelu) v = fmaxf(v, 0.0f);
        Abuf[addr] = v;
    }
}

// ---------------------------------------------------------------------------
// K9: head on quadrant only: fc1(32->128, WMMA) + relu + fc2(128->1, VALU),
//     broadcast result to 3 output channels.
// ---------------------------------------------------------------------------
__global__ void fno_head(const float* __restrict__ Abuf,
                         const float* __restrict__ fc1_w,
                         const float* __restrict__ fc1_b,
                         const float* __restrict__ fc2_w,
                         const float* __restrict__ fc2_b,
                         float* __restrict__ out) {
    __shared__ float hid[16 * 128];
    __shared__ float part[32];
    int lane = threadIdx.x & 31;
    int m = lane & 15, half = lane >> 4, n = lane & 15;
    int q0 = blockIdx.x * 16;

    int qp = q0 + m;
    int bb = qp >> 14;
    int rem = qp & 16383;
    int qi = rem >> 7, qj = rem & 127;
    int hw = (NS - 1 + qi) * 256 + (NS - 1 + qj);
    v16h a;
#pragma unroll
    for (int e = 0; e < 16; ++e) {
        int cch = fragA_k(e, half);
        a[e] = (_Float16)Abuf[(size_t)bb * CHW2 + (size_t)cch * PL + hw];
    }
#pragma unroll
    for (int nt = 0; nt < 8; ++nt) {
        v16h b;
#pragma unroll
        for (int e = 0; e < 16; ++e) {
            int cch = fragB_k(e, half);
            int d = nt * 16 + n;
            b[e] = (_Float16)fc1_w[cch * 128 + d];
        }
        v8f c = vzero8();
        c = WMMA(a, b, c);
#pragma unroll
        for (int r = 0; r < 8; ++r) {
            int row = r + 8 * half;
            int d = nt * 16 + n;
            hid[row * 128 + d] = fmaxf(c[r] + fc1_b[d], 0.0f);
        }
    }
    __syncthreads();
    // fc2: two lanes per pixel, 64 elements each
    {
        int px = lane & 15, hs = lane >> 4;
        float s = 0.0f;
#pragma unroll
        for (int t = 0; t < 64; ++t)
            s += hid[px * 128 + hs * 64 + t] * fc2_w[hs * 64 + t];
        part[lane] = s;
    }
    __syncthreads();
    if (lane < 16) {
        float v = part[lane] + part[lane + 16] + fc2_b[0];
        int qp2 = q0 + lane;
        int b2 = qp2 >> 14;
        int rem2 = qp2 & 16383;
        int qi2 = rem2 >> 7, qj2 = rem2 & 127;
#pragma unroll
        for (int c = 0; c < 3; ++c)
            out[(((size_t)b2 * 3 + c) * NS + qi2) * NS + qj2] = v;
    }
}

// ---------------------------------------------------------------------------
// Host orchestration
// ---------------------------------------------------------------------------
extern "C" void kernel_launch(void* const* d_in, const int* in_sizes, int n_in,
                              void* d_out, int out_size, void* d_ws, size_t ws_size,
                              hipStream_t stream) {
    const float* x      = (const float*)d_in[0];
    const float* fc0_w  = (const float*)d_in[1];
    const float* fc0_b  = (const float*)d_in[2];
    const float* spec_w = (const float*)d_in[3];
    const float* conv_w = (const float*)d_in[4];
    const float* conv_b = (const float*)d_in[5];
    const float* bn_g   = (const float*)d_in[6];
    const float* bn_b   = (const float*)d_in[7];
    const float* fc1_w  = (const float*)d_in[8];
    const float* fc1_b  = (const float*)d_in[9];
    const float* fc2_w  = (const float*)d_in[10];
    const float* fc2_b  = (const float*)d_in[11];
    float* out = (float*)d_out;

    float* W = (float*)d_ws;
    const size_t BIG = (size_t)NB * CHW2;                // 33,423,360
    float* bufA = W;                                      // h
    float* bufS = bufA + BIG;                             // x1 + x2 (pre-BN)
    float* Yf   = bufS + BIG;                             // 130560 x 32
    float* Zf   = Yf + (size_t)NB * NC * NH * YF_STR;     // 512*48 x 16
    float* Zm   = Zf + (size_t)NB * NC * 2 * NK1 * ZV_STR;
    float* Yi   = Zm + (size_t)NB * NC * 2 * NK1 * ZV_STR; // 512*256 x 32
    float* DWF  = Yi + (size_t)NB * NC * YI_ROWS * 32;
    float* DHF  = DWF + 256 * 32;
    float* DHI  = DHF + 48 * 512;
    float* DWI  = DHI + 256 * 64;
    float* stats = DWI + 32 * 256;

    fno_init_tables<<<96, 256, 0, stream>>>(DWF, DHF, DHI, DWI);
    fno_mirror_fc0<<<4096, 256, 0, stream>>>(x, fc0_w, fc0_b, bufA);

    const int rowTiles = (NB * NC * NH) / 16;   // 8160
    const int pixTiles = (NB * HW + 15) / 16;   // 65025
    for (int blk = 0; blk < 4; ++blk) {
        fno_dft_fwd_w<<<rowTiles, 32, 0, stream>>>(bufA, DWF, Yf);
        fno_dft_fwd_h<<<NB * NC, 32, 0, stream>>>(Yf, DHF, Zf);
        fno_mix<<<NK1 * NM2, 32, 0, stream>>>(Zf, spec_w, Zm, blk);
        fno_dft_inv_h<<<NB * NC, 32, 0, stream>>>(Zm, DHI, Yi);
        fno_dft_inv_w<<<rowTiles, 32, 0, stream>>>(Yi, DWI, bufS);
        fno_conv_add<<<pixTiles, 32, 0, stream>>>(bufA, conv_w, conv_b, bufS, blk);
        fno_zero<<<1, 64, 0, stream>>>(stats, 2 * NC);
        fno_bn_stats<<<NC * 8, 256, 0, stream>>>(bufS, stats);
        fno_bn_apply<<<8192, 256, 0, stream>>>(bufS, stats, bn_g, bn_b, bufA, blk, blk < 3 ? 1 : 0);
    }
    fno_head<<<(NB * NS * NS) / 16, 32, 0, stream>>>(bufA, fc1_w, fc1_b, fc2_w, fc2_b, out);
    (void)in_sizes; (void)n_in; (void)out_size; (void)ws_size;
}
